// kaux_real_62474594288120
// MI455X (gfx1250) — matrix operand order repaired and checked by reading, not compiled, not run on hardware
//
#include <hip/hip_runtime.h>
#include <hip/hip_bf16.h>
#include <cstdint>
#include <cstddef>

// GRU dims (fixed by the reference)
#define B_  64
#define T_  256
#define F_  256
#define U_  1024
#define G3_ 3072            // 3*U
#define M_  (B_ * T_)       // 16384 rows for the input-projection GEMM

typedef __attribute__((ext_vector_type(16))) __bf16 v16bf;
typedef __attribute__((ext_vector_type(8)))  float  v8f;

__device__ __forceinline__ float sigmoid_f(float x) {
    return 1.0f / (1.0f + __expf(-x));
}

// Per-lane async copy of 16 bytes global -> LDS (ASYNCcnt tracked, no VGPR trip).
__device__ __forceinline__ void async_g2lds_b128(uint32_t lds_byte_addr, const void* gaddr) {
    asm volatile("global_load_async_to_lds_b128 %0, %1, off"
                 :: "v"(lds_byte_addr), "v"(gaddr) : "memory");
}
__device__ __forceinline__ void wait_asynccnt0() {
    asm volatile("s_wait_asynccnt 0" ::: "memory");
}

// ---------------------------------------------------------------------------
// WMMA fragment loaders (wave32, v_wmma_f32_16x16x32_bf16; contiguous k order, same for A and B)
//   A (16x32, MxK): lanes 0-15 -> M=lane, elems = K+0..15
//                   lanes 16-31 -> M=lane-16, elems = K+16..31
//   B (32x16, KxN): lanes 0-15 -> N=lane, elems = K+0..15 ; lanes 16-31 -> K+16..31
//   C/D (16x16)   : N = lane&15, M = vgpr + 8*(lane>=16)
// ---------------------------------------------------------------------------
__device__ __forceinline__ v16bf frag_a_bf16(const __bf16* base, int ldk, int row0, int k0, int lane) {
    const int r  = row0 + (lane & 15);
    const int kb = k0 + ((lane >> 4) << 4);
    const __bf16* p = base + (size_t)r * ldk + kb;
    v16bf a;
#pragma unroll
    for (int i = 0; i < 8; ++i) { a[i] = p[i]; a[8 + i] = p[8 + i]; }
    return a;
}

// ---------------------------------------------------------------------------
// Small conversion / copy kernels
// ---------------------------------------------------------------------------
__global__ void cast_f32_to_bf16(const float* __restrict__ in, __bf16* __restrict__ out, size_t n) {
    size_t i = (size_t)blockIdx.x * blockDim.x + threadIdx.x;
    if (i < n) out[i] = (__bf16)in[i];
}

// in: [K, N] f32 row-major  ->  out: [N, K] bf16 row-major
__global__ void transpose_f32_to_bf16(const float* __restrict__ in, __bf16* __restrict__ out, int K, int N) {
    int k = blockIdx.x * 32 + threadIdx.x;
    int n = blockIdx.y * 8 + threadIdx.y;
    if (k < K && n < N) out[(size_t)n * K + k] = (__bf16)in[(size_t)k * N + n];
}

__global__ void copy_f32(const float* __restrict__ in, float* __restrict__ out, size_t n) {
    size_t i = (size_t)blockIdx.x * blockDim.x + threadIdx.x;
    if (i < n) out[i] = in[i];
}

// ---------------------------------------------------------------------------
// gx = xb @ W^T(bf16, [3U,K]) + b0 ; xb [M,K] bf16 ; gx [M, 3U] f32
// block = 128 threads = 4 waves; wave -> 16(M) x 64(N) strip (4 acc tiles).
// The shared 64-col B slab per K-step is staged into LDS with async
// global->LDS b128 copies (256 x 16B quanta = 128 threads x 2).
// grid = (G3/64, M/64)
// ---------------------------------------------------------------------------
__global__ void __launch_bounds__(128)
gru_gx_gemm(const __bf16* __restrict__ xb, const __bf16* __restrict__ wt,
            const float* __restrict__ bias0, float* __restrict__ gx, int K)
{
    __shared__ __bf16 Bs[64][32];      // [col][k], k contiguous
    const int tid  = threadIdx.x;
    const int lane = tid & 31;
    const int wave = tid >> 5;
    const int n0 = blockIdx.x * 64;
    const int m0 = (blockIdx.y * 4 + wave) * 16;
    const uint32_t lds_base = (uint32_t)(uintptr_t)(&Bs[0][0]);

    const v8f vz = {0.f, 0.f, 0.f, 0.f, 0.f, 0.f, 0.f, 0.f};
    v8f acc[4] = {vz, vz, vz, vz};

    for (int k0 = 0; k0 < K; k0 += 32) {
        __syncthreads();               // previous-iteration LDS reads complete
        {
            const int task0 = tid * 2;
#pragma unroll
            for (int s = 0; s < 2; ++s) {
                const int task = task0 + s;
                const int col = task >> 2;      // 0..63
                const int oct = task & 3;       // 8 bf16 (16B) per quantum
                const __bf16* gsrc = wt + (size_t)(n0 + col) * K + k0 + oct * 8;
                const uint32_t ldst = lds_base + (uint32_t)((col * 32 + oct * 8) * 2);
                async_g2lds_b128(ldst, gsrc);
            }
        }
        if (k0 + 32 < K)
            __builtin_prefetch(xb + (size_t)(m0 + (lane & 15)) * K + k0 + 32, 0, 3);

        v16bf a = frag_a_bf16(xb, K, m0, k0, lane);

        wait_asynccnt0();
        __syncthreads();

        const int cl = lane & 15;
        const int kb = (lane >> 4) << 4;
#pragma unroll
        for (int t = 0; t < 4; ++t) {
            v16bf b;
#pragma unroll
            for (int i = 0; i < 16; ++i) b[i] = Bs[16 * t + cl][kb + i];
            acc[t] = __builtin_amdgcn_wmma_f32_16x16x32_bf16(
                false, a, false, b, (short)0, acc[t], false, false);
        }
    }

    const int cl = lane & 15;
    const int rb = m0 + ((lane >> 4) << 3);
#pragma unroll
    for (int t = 0; t < 4; ++t) {
        const int c = n0 + 16 * t + cl;
        const float bv = bias0[c];
#pragma unroll
        for (int v = 0; v < 8; ++v)
            gx[(size_t)(rb + v) * G3_ + c] = acc[t][v] + bv;
    }
}

// ---------------------------------------------------------------------------
// One recurrence step (latency-critical; 768 sequential launches total):
//   rec = h_in @ U^T + b1 ; z=sig(gx_z+rec_z); r=sig(gx_r+rec_r);
//   hh=tanh(gx_h + r*rec_h); h_out = z*h_in + (1-z)*hh ; y_t = h_out
// grid = U/16 blocks (one 16-wide j slice). Block = 256 threads = 8 waves:
//   wave w -> batch-row tile (w&3), K-half (w>>2) of the 1024 reduction.
// K split halves the dependent-WMMA chain (16 instead of 32) and doubles
// wave parallelism; partials combined through LDS, waves 0-3 run the fused
// gate epilogue. A-matrix comes from a bf16 shadow of h (no per-step cvt).
// Both K-half B slabs (3 gates x 16 cols x 32 k x 2) staged per iteration
// via async global->LDS b128 copies (384 quanta = threads 0-191 x 2).
// ---------------------------------------------------------------------------
__global__ void __launch_bounds__(256)
gru_step(const __bf16* __restrict__ hb_in,   // bf16 shadow of h   [64,1024]
         const float* __restrict__ h_in,     // f32 h              [64,1024]
         const __bf16* __restrict__ ut,      // U transposed bf16  [3072,1024]
         const float* __restrict__ bias1,    // recurrent bias     [3072]
         const float* __restrict__ gx_t,     // gx at time t (row stride T*3U)
         float* __restrict__ h_out, __bf16* __restrict__ hb_out,
         float* __restrict__ y_t)            // y at time t (row stride T*U)
{
    __shared__ __bf16 Bs[2][3][16][32];      // [khalf][gate][col][k]
    __shared__ float  Red[4][3][16][16];     // [mtile][gate][row][col] partials

    const int tid   = threadIdx.x;
    const int lane  = tid & 31;
    const int wave  = tid >> 5;              // 0..7
    const int mt    = wave & 3;              // batch-row tile
    const int khalf = wave >> 2;             // 0 or 1
    const int j0 = blockIdx.x * 16;          // output column slice
    const int m0 = mt * 16;
    const uint32_t lds_base = (uint32_t)(uintptr_t)(&Bs[0][0][0][0]);

    const v8f vz = {0.f, 0.f, 0.f, 0.f, 0.f, 0.f, 0.f, 0.f};
    v8f az = vz, ar = vz, ah = vz;

    for (int k0 = 0; k0 < U_ / 2; k0 += 32) {
        __syncthreads();                     // previous-iteration LDS reads done
        if (tid < 192) {                     // 384 quanta of 16B, 2 per thread
            const int task0 = tid * 2;
#pragma unroll
            for (int s = 0; s < 2; ++s) {
                const int task = task0 + s;
                const int kh   = task >= 192;
                const int rem  = task - 192 * kh;
                const int row  = rem >> 2;   // 0..47 = gate*16 + col
                const int g    = row >> 4;
                const int c    = row & 15;
                const int oct  = rem & 3;
                const __bf16* gsrc =
                    ut + (size_t)(g * U_ + j0 + c) * U_ + kh * (U_ / 2) + k0 + oct * 8;
                const uint32_t ldst =
                    lds_base + (uint32_t)(((kh * 48 + row) * 32 + oct * 8) * 2);
                async_g2lds_b128(ldst, gsrc);
            }
        }

        v16bf a = frag_a_bf16(hb_in, U_, m0, khalf * (U_ / 2) + k0, lane);

        wait_asynccnt0();
        __syncthreads();

        const int c  = lane & 15;
        const int kb = (lane >> 4) << 4;
        v16bf bz, br, bh;
#pragma unroll
        for (int i = 0; i < 16; ++i) {
            bz[i] = Bs[khalf][0][c][kb + i];
            br[i] = Bs[khalf][1][c][kb + i];
            bh[i] = Bs[khalf][2][c][kb + i];
        }
        az = __builtin_amdgcn_wmma_f32_16x16x32_bf16(false, a, false, bz, (short)0, az, false, false);
        ar = __builtin_amdgcn_wmma_f32_16x16x32_bf16(false, a, false, br, (short)0, ar, false, false);
        ah = __builtin_amdgcn_wmma_f32_16x16x32_bf16(false, a, false, bh, (short)0, ah, false, false);
    }

    const int cl = lane & 15;
    const int rb = (lane >> 4) << 3;         // 0 or 8: row offset within tile

    // Combine K-half partials through LDS.
    if (khalf == 1) {
#pragma unroll
        for (int v = 0; v < 8; ++v) {
            Red[mt][0][rb + v][cl] = az[v];
            Red[mt][1][rb + v][cl] = ar[v];
            Red[mt][2][rb + v][cl] = ah[v];
        }
    }
    __syncthreads();
    if (khalf == 0) {
        const int j = j0 + cl;
        const float bzc = bias1[j];
        const float brc = bias1[U_ + j];
        const float bhc = bias1[2 * U_ + j];
#pragma unroll
        for (int v = 0; v < 8; ++v) {
            const int b = m0 + rb + v;
            const float* gp = gx_t + (size_t)b * ((size_t)T_ * G3_);
            const float rz = az[v] + Red[mt][0][rb + v][cl] + bzc;
            const float rr = ar[v] + Red[mt][1][rb + v][cl] + brc;
            const float rh = ah[v] + Red[mt][2][rb + v][cl] + bhc;
            const float z  = sigmoid_f(gp[j] + rz);
            const float r  = sigmoid_f(gp[U_ + j] + rr);
            const float hh = tanhf(gp[2 * U_ + j] + r * rh);
            const float ho = h_in[(size_t)b * U_ + j];
            const float hn = z * ho + (1.0f - z) * hh;
            h_out[(size_t)b * U_ + j]  = hn;
            hb_out[(size_t)b * U_ + j] = (__bf16)hn;
            y_t[(size_t)b * ((size_t)T_ * U_) + j] = hn;
        }
    }
}

// ---------------------------------------------------------------------------
// Host orchestration
// ---------------------------------------------------------------------------
extern "C" void kernel_launch(void* const* d_in, const int* in_sizes, int n_in,
                              void* d_out, int out_size, void* d_ws, size_t ws_size,
                              hipStream_t stream)
{
    // setup_inputs order: inputs, {state_i, W_i, U_i, b_i} for i=0..2
    const float* x_in   = (const float*)d_in[0];
    const float* st[3]  = {(const float*)d_in[1], (const float*)d_in[5], (const float*)d_in[9]};
    const float* Wm[3]  = {(const float*)d_in[2], (const float*)d_in[6], (const float*)d_in[10]};
    const float* Urm[3] = {(const float*)d_in[3], (const float*)d_in[7], (const float*)d_in[11]};
    const float* bm[3]  = {(const float*)d_in[4], (const float*)d_in[8], (const float*)d_in[12]};

    char* ws = (char*)d_ws;
    size_t off = 0;
    auto take = [&](size_t bytes) -> char* {
        char* p = ws + off;
        off = (off + bytes + 255) & ~(size_t)255;
        return p;
    };

    __bf16* xb  = (__bf16*)take((size_t)M_ * U_ * 2);        // bf16 layer input
    __bf16* wt0 = (__bf16*)take((size_t)G3_ * F_ * 2);       // W transposed, bf16
    __bf16* wt1 = (__bf16*)take((size_t)G3_ * U_ * 2);
    __bf16* wt2 = (__bf16*)take((size_t)G3_ * U_ * 2);
    __bf16* ut0 = (__bf16*)take((size_t)G3_ * U_ * 2);       // U transposed, bf16
    __bf16* ut1 = (__bf16*)take((size_t)G3_ * U_ * 2);
    __bf16* ut2 = (__bf16*)take((size_t)G3_ * U_ * 2);
    float*  gx   = (float*)take((size_t)M_ * G3_ * 4);       // input projections
    float*  ybuf = (float*)take((size_t)M_ * U_ * 4);        // inter-layer activations
    float*  h0   = (float*)take((size_t)B_ * U_ * 4);        // h double-buffer (f32)
    float*  h1   = (float*)take((size_t)B_ * U_ * 4);
    __bf16* hb0  = (__bf16*)take((size_t)B_ * U_ * 2);       // h double-buffer (bf16)
    __bf16* hb1  = (__bf16*)take((size_t)B_ * U_ * 2);
    __bf16* stb  = (__bf16*)take((size_t)B_ * U_ * 2);       // bf16 initial state

    __bf16* wt[3] = {wt0, wt1, wt2};
    __bf16* ut[3] = {ut0, ut1, ut2};

    // One-time weight transpose + bf16 conversion
    {
        dim3 blk(32, 8);
        dim3 gW0((F_ + 31) / 32, (G3_ + 7) / 8);
        dim3 gU((U_ + 31) / 32, (G3_ + 7) / 8);
        transpose_f32_to_bf16<<<gW0, blk, 0, stream>>>(Wm[0],  wt0, F_, G3_);
        transpose_f32_to_bf16<<<gU,  blk, 0, stream>>>(Wm[1],  wt1, U_, G3_);
        transpose_f32_to_bf16<<<gU,  blk, 0, stream>>>(Wm[2],  wt2, U_, G3_);
        transpose_f32_to_bf16<<<gU,  blk, 0, stream>>>(Urm[0], ut0, U_, G3_);
        transpose_f32_to_bf16<<<gU,  blk, 0, stream>>>(Urm[1], ut1, U_, G3_);
        transpose_f32_to_bf16<<<gU,  blk, 0, stream>>>(Urm[2], ut2, U_, G3_);
    }

    float* out_y = (float*)d_out;                 // [B,T,U]
    float* out_s = out_y + (size_t)M_ * U_;       // s0,s1,s2 each [B,U]

    for (int layer = 0; layer < 3; ++layer) {
        const int K = (layer == 0) ? F_ : U_;
        const float* xsrc = (layer == 0) ? x_in : ybuf;

        {   // layer input -> bf16
            size_t n = (size_t)M_ * K;
            cast_f32_to_bf16<<<(int)((n + 255) / 256), 256, 0, stream>>>(xsrc, xb, n);
        }
        // bf16 shadow of the initial state
        cast_f32_to_bf16<<<(B_ * U_ + 255) / 256, 256, 0, stream>>>(
            st[layer], stb, (size_t)B_ * U_);

        // gx = xb @ W^T + b0
        gru_gx_gemm<<<dim3(G3_ / 64, M_ / 64), 128, 0, stream>>>(xb, wt[layer], bm[layer], gx, K);

        // sequential scan over time (stream order enforces h dependency)
        float* yt_base = (layer == 2) ? out_y : ybuf;
        const float* b1 = bm[layer] + G3_;
        for (int t = 0; t < T_; ++t) {
            const float*  hin  = (t == 0) ? st[layer] : ((t & 1) ? h0 : h1);
            const __bf16* hbin = (t == 0) ? stb       : ((t & 1) ? hb0 : hb1);
            float*  hout  = (t & 1) ? h1 : h0;
            __bf16* hbout = (t & 1) ? hb1 : hb0;
            gru_step<<<dim3(U_ / 16), 256, 0, stream>>>(
                hbin, hin, ut[layer], b1, gx + (size_t)t * G3_,
                hout, hbout, yt_base + (size_t)t * U_);
        }
        // T_-1 = 255 is odd -> last write went to h1
        copy_f32<<<(B_ * U_ + 255) / 256, 256, 0, stream>>>(
            h1, out_s + (size_t)layer * B_ * U_, (size_t)B_ * U_);
    }

    (void)in_sizes; (void)n_in; (void)out_size; (void)ws_size;
}
